// SPIntraAttModule_33346126086739
// MI455X (gfx1250) — matrix operand order repaired and checked by reading, not compiled, not run on hardware
//
#include <hip/hip_runtime.h>
#include <hip/hip_fp16.h>

typedef __attribute__((ext_vector_type(16))) _Float16 v16h;
typedef __attribute__((ext_vector_type(8)))  _Float16 v8h;
typedef __attribute__((ext_vector_type(8)))  float    v8f;

#define NB    8
#define NC    128
#define NHW   16384
#define NS    256
#define NTOP  128
#define NHEAD 4
#define NHD   32
static constexpr float kScale = 0.17677669529663687f;   // 32^-0.5
static constexpr float kEps   = 1e-6f;

struct uint4x2 { uint4 lo; uint4 hi; };

static __device__ __forceinline__ v16h make_v16h(uint4 lo, uint4 hi) {
    uint4x2 t{lo, hi};
    return __builtin_bit_cast(v16h, t);
}
static __device__ __forceinline__ uint4 pack8h(v8f a) {
    v8h h;
    #pragma unroll
    for (int i = 0; i < 8; ++i) h[i] = (_Float16)a[i];
    return __builtin_bit_cast(uint4, h);
}

// ---------------------------------------------------------------- utilities
__global__ __launch_bounds__(256) void zero_counts_kernel(float* __restrict__ counts) {
    int i = blockIdx.x * 256 + threadIdx.x;   // NB*NHW = 131072
    counts[i] = 0.0f;
}

// convert f32 weights (wq|wk|wv) to f16, row-major [mat][o][c]
__global__ __launch_bounds__(256) void wcvt_kernel(const float* __restrict__ wq,
                                                   const float* __restrict__ wk,
                                                   const float* __restrict__ wv,
                                                   _Float16* __restrict__ wf) {
    int i = blockIdx.x * 256 + threadIdx.x;   // 3*16384
    int sel = i >> 14, off = i & 16383;
    const float* src = (sel == 0) ? wq : (sel == 1) ? wk : wv;
    wf[i] = (_Float16)src[off];
}

// ------------------------------------------- LayerNorm -> f16, pixel-major
__global__ __launch_bounds__(256) void ln_kernel(const float* __restrict__ x,
                                                 const float* __restrict__ gamma,
                                                 const float* __restrict__ beta,
                                                 _Float16* __restrict__ xn) {
    int gid = blockIdx.x * 256 + threadIdx.x;   // NB*NHW threads
    int b = gid >> 14;
    int n = gid & (NHW - 1);
    const float* xp = x + (size_t)b * NC * NHW + n;

    float vals[NC];
    float s = 0.f, sq = 0.f;
    #pragma unroll
    for (int c = 0; c < NC; ++c) {
        float v = xp[(size_t)c * NHW];
        vals[c] = v; s += v; sq += v * v;
    }
    float mu  = s * (1.0f / NC);
    float var = sq * (1.0f / NC) - mu * mu;
    float inv = rsqrtf(var + kEps);

    _Float16* op = xn + ((size_t)b * NHW + n) * NC;   // pixel-major row
    #pragma unroll
    for (int c0 = 0; c0 < NC; c0 += 8) {
        v8h pk;
        #pragma unroll
        for (int j = 0; j < 8; ++j)
            pk[j] = (_Float16)((vals[c0 + j] - mu) * inv * gamma[c0 + j] + beta[c0 + j]);
        *(uint4*)(op + c0) = __builtin_bit_cast(uint4, pk);
    }
}

// --------------------------------------------------- fused Q/K/V WMMA GEMM
// out[px][ch] = sum_c W[ch][c] * xn[px][c] ; all operands b128 loads.
__global__ __launch_bounds__(256) void qkv_kernel(const _Float16* __restrict__ xn,
                                                  const _Float16* __restrict__ wf,
                                                  _Float16* __restrict__ qo,
                                                  _Float16* __restrict__ ko,
                                                  _Float16* __restrict__ vo,
                                                  float* __restrict__ vout) {
    int blk  = blockIdx.x;                      // NB * (NHW/128) = 1024
    int b    = blk >> 7;
    int tile = blk & 127;
    int tid  = threadIdx.x;
    int wave = tid >> 5, lane = tid & 31;
    int half = lane >> 4, l15 = lane & 15;
    int mrow = wave * 16 + l15;                 // output-channel row for A

    // A operands for all three matrices (b128 loads of f16 weights)
    v16h A[3][4];
    #pragma unroll
    for (int mat = 0; mat < 3; ++mat) {
        const _Float16* W = wf + mat * (NC * NC) + mrow * NC;
        #pragma unroll
        for (int ks = 0; ks < 4; ++ks) {
            const uint4* p0 = (const uint4*)(W + ks * 32 + half * 8);
            const uint4* p1 = (const uint4*)(W + ks * 32 + 16 + half * 8);
            A[mat][ks] = make_v16h(*p0, *p1);
        }
    }

    for (int nt = 0; nt < 8; ++nt) {
        int px = tile * 128 + nt * 16 + l15;
        const _Float16* xrow = xn + ((size_t)b * NHW + px) * NC;

        v16h Bv[4];
        #pragma unroll
        for (int ks = 0; ks < 4; ++ks) {
            int kb = ks * 32 + half * 8;
            Bv[ks] = make_v16h(*(const uint4*)(xrow + kb),
                               *(const uint4*)(xrow + kb + 16));
        }

        size_t orow = ((size_t)b * NHW + px) * NC + wave * 16 + 8 * half;
        #pragma unroll
        for (int mat = 0; mat < 3; ++mat) {
            v8f acc = {};
            #pragma unroll
            for (int ks = 0; ks < 4; ++ks)
                acc = __builtin_amdgcn_wmma_f32_16x16x32_f16(
                        false, A[mat][ks], false, Bv[ks], (short)0, acc, false, false);
            if (mat == 0)      *(uint4*)(qo + orow) = pack8h(acc);
            else if (mat == 1) *(uint4*)(ko + orow) = pack8h(acc);
            else {
                *(uint4*)(vo + orow) = pack8h(acc);
                // d_out := V  (channel-major output layout, scalar stores)
                #pragma unroll
                for (int r = 0; r < 8; ++r) {
                    int ch = wave * 16 + 8 * half + r;
                    vout[((size_t)b * NC + ch) * NHW + px] = acc[r];
                }
            }
        }
    }
}

// --------------------------------------------------- radix top-128 select
__global__ __launch_bounds__(256) void topk_kernel(const float* __restrict__ aff,
                                                   int* __restrict__ idxo,
                                                   float* __restrict__ counts) {
    __shared__ int hist[2048];
    __shared__ int sh_pivot, sh_need;
    __shared__ int cHigh, cTie;

    int row = blockIdx.x;                       // b*NS + s  (2048 rows)
    int b   = row >> 8;
    const float* rp = aff + (size_t)row * NHW;
    int tid = threadIdx.x;

    auto keyOf = [](float f) -> unsigned {
        unsigned u = __float_as_uint(f);
        return u ^ ((u >> 31) ? 0xFFFFFFFFu : 0x80000000u);
    };

    const int      shifts[3]  = { 21, 10, 0 };
    const unsigned masksel[3] = { 0x7FFu, 0x7FFu, 0x3FFu };

    unsigned prefix = 0;
    int need = NTOP;
    for (int level = 0; level < 3; ++level) {
        int nb = (level == 2) ? 1024 : 2048;
        for (int i = tid; i < nb; i += 256) hist[i] = 0;
        __syncthreads();
        for (int i = tid; i < NHW; i += 256) {
            unsigned key = keyOf(rp[i]);
            if (level == 0 || (key >> shifts[level - 1]) == prefix) {
                unsigned bin = (key >> shifts[level]) & masksel[level];
                atomicAdd(&hist[bin], 1);
            }
        }
        __syncthreads();
        if (tid == 0) {
            int cum = 0, piv = 0, nd = need;
            for (int bn = nb - 1; bn >= 0; --bn) {
                cum += hist[bn];
                if (cum >= need) { piv = bn; nd = need - (cum - hist[bn]); break; }
            }
            sh_pivot = piv; sh_need = nd;
        }
        __syncthreads();
        prefix = (prefix << ((level == 2) ? 10 : 11)) | (unsigned)sh_pivot;
        need   = sh_need;
        __syncthreads();
    }

    if (tid == 0) { cHigh = 0; cTie = 0; }
    __syncthreads();
    unsigned th = prefix;            // exact 32-bit threshold key
    int nTies = need;
    int nHigh = NTOP - nTies;
    int* op = idxo + row * NTOP;
    for (int i = tid; i < NHW; i += 256) {
        unsigned key = keyOf(rp[i]);
        int pos = -1;
        if (key > th) pos = atomicAdd(&cHigh, 1);
        else if (key == th) {
            int j = atomicAdd(&cTie, 1);
            if (j < nTies) pos = nHigh + j;
        }
        if (pos >= 0) {
            op[pos] = i;
            unsafeAtomicAdd(&counts[b * NHW + i], 1.0f);
        }
    }
}

// --------------------------------------------------- per-(b,s,h) attention
__global__ __launch_bounds__(256) void attn_kernel(const _Float16* __restrict__ qws,
                                                   const _Float16* __restrict__ kws,
                                                   const _Float16* __restrict__ vws,
                                                   const int* __restrict__ idx,
                                                   float* __restrict__ out) {
    __shared__ int sidx[128];
    __shared__ __align__(16) _Float16 Qg[128][40];   // [t][c]  A of QK^T
    __shared__ __align__(16) _Float16 Kg[128][40];   // [u][c]  B of QK^T (transposed form)
    __shared__ __align__(16) _Float16 Vt[32][136];   // [d][u]  B of PV   (transposed form)
    __shared__ __align__(16) _Float16 Sb[128][136];  // [t][u]  scores / probs

    int blk = blockIdx.x;                // NB*NS*NHEAD = 8192
    int h = blk & 3, s = (blk >> 2) & 255, b = blk >> 10;
    int tid = threadIdx.x;

    if (tid < 128) sidx[tid] = idx[(size_t)(b * NS + s) * NTOP + tid] & (NHW - 1);
    __syncthreads();

    // gather: 16B chunks per (t, 8-channel group); V transposed on store
    for (int i = tid; i < 512; i += 256) {
        int t = i >> 2, c0 = (i & 3) * 8;
        size_t g = ((size_t)b * NHW + sidx[t]) * NC + h * NHD + c0;
        *(uint4*)&Qg[t][c0] = *(const uint4*)(qws + g);
        *(uint4*)&Kg[t][c0] = *(const uint4*)(kws + g);
        v8h dv = *(const v8h*)(vws + g);
        #pragma unroll
        for (int j = 0; j < 8; ++j) Vt[c0 + j][t] = dv[j];
    }
    __syncthreads();

    int wave = tid >> 5, lane = tid & 31, half = lane >> 4, l15 = lane & 15;

    // S = scale * Q K^T   (M=N=128, K=32 -> one WMMA per 16x16 tile)
    {
        int t = wave * 16 + l15;
        v16h A = make_v16h(*(const uint4*)&Qg[t][half * 8],
                           *(const uint4*)&Qg[t][16 + half * 8]);
        for (int nt = 0; nt < 8; ++nt) {
            int u = nt * 16 + l15;
            v16h Bv = make_v16h(*(const uint4*)&Kg[u][half * 8],
                                *(const uint4*)&Kg[u][16 + half * 8]);
            v8f acc = {};
            acc = __builtin_amdgcn_wmma_f32_16x16x32_f16(
                    false, A, false, Bv, (short)0, acc, false, false);
            #pragma unroll
            for (int r = 0; r < 8; ++r)
                Sb[wave * 16 + r + 8 * half][u] = (_Float16)(acc[r] * kScale);
        }
    }
    __syncthreads();

    // row softmax, vectorized 16B accesses (one thread per row)
    if (tid < 128) {
        float mx = -1e30f;
        for (int u0 = 0; u0 < 128; u0 += 8) {
            v8h c = *(const v8h*)&Sb[tid][u0];
            #pragma unroll
            for (int j = 0; j < 8; ++j) mx = fmaxf(mx, (float)c[j]);
        }
        float sm = 0.f;
        for (int u0 = 0; u0 < 128; u0 += 8) {
            v8h c = *(const v8h*)&Sb[tid][u0];
            #pragma unroll
            for (int j = 0; j < 8; ++j) sm += __expf((float)c[j] - mx);
        }
        float inv = 1.0f / sm;
        for (int u0 = 0; u0 < 128; u0 += 8) {
            v8h c = *(const v8h*)&Sb[tid][u0];
            #pragma unroll
            for (int j = 0; j < 8; ++j)
                c[j] = (_Float16)(__expf((float)c[j] - mx) * inv);
            *(v8h*)&Sb[tid][u0] = c;
        }
    }
    __syncthreads();

    // O = P @ V  (M=128, N=32, K=128) then scatter-add into d_out
    {
        int t = wave * 16 + l15;
        for (int nt = 0; nt < 2; ++nt) {
            int d = nt * 16 + l15;
            v8f acc = {};
            #pragma unroll
            for (int ks = 0; ks < 4; ++ks) {
                v16h A = make_v16h(*(const uint4*)&Sb[t][ks * 32 + half * 8],
                                   *(const uint4*)&Sb[t][ks * 32 + 16 + half * 8]);
                v16h Bv = make_v16h(*(const uint4*)&Vt[d][ks * 32 + half * 8],
                                    *(const uint4*)&Vt[d][ks * 32 + 16 + half * 8]);
                acc = __builtin_amdgcn_wmma_f32_16x16x32_f16(
                        false, A, false, Bv, (short)0, acc, false, false);
            }
            #pragma unroll
            for (int r = 0; r < 8; ++r) {
                int tt = wave * 16 + r + 8 * half;
                int ch = h * NHD + nt * 16 + l15;
                int px = sidx[tt];
                unsafeAtomicAdd(&out[((size_t)b * NC + ch) * NHW + px], acc[r]);
            }
        }
    }
}

// ---------------------------------------------------------------- finalize
__global__ __launch_bounds__(256) void finalize_kernel(float* __restrict__ out,
                                                       const float* __restrict__ counts) {
    size_t i = (size_t)blockIdx.x * 256 + threadIdx.x;    // NB*NC*NHW
    int b  = (int)(i >> 21);                              // / (NC*NHW)
    int px = (int)(i & (NHW - 1));
    out[i] = out[i] / (1.0f + counts[b * NHW + px]);
}

// ---------------------------------------------------------------- launcher
extern "C" void kernel_launch(void* const* d_in, const int* in_sizes, int n_in,
                              void* d_out, int out_size, void* d_ws, size_t ws_size,
                              hipStream_t stream) {
    (void)in_sizes; (void)n_in; (void)out_size; (void)ws_size;
    const float* x     = (const float*)d_in[0];
    const float* aff   = (const float*)d_in[1];
    // d_in[2] = num_spixels (compile-time NS)
    const float* wq    = (const float*)d_in[3];
    const float* wk    = (const float*)d_in[4];
    const float* wv    = (const float*)d_in[5];
    const float* gamma = (const float*)d_in[6];
    const float* beta  = (const float*)d_in[7];
    float* out = (float*)d_out;

    char* ws = (char*)d_ws;
    const size_t f16buf = (size_t)NB * NC * NHW * sizeof(_Float16);   // 32 MiB
    _Float16* xn   = (_Float16*)(ws);                  // [b][n][c] pixel-major
    _Float16* qws  = (_Float16*)(ws + 1 * f16buf);     // [b][n][c]
    _Float16* kws  = (_Float16*)(ws + 2 * f16buf);     // [b][n][c]
    _Float16* vws  = (_Float16*)(ws + 3 * f16buf);     // [b][n][c]
    int*      idx  = (int*)     (ws + 4 * f16buf);
    float*  counts = (float*)   (ws + 4 * f16buf + (size_t)NB * NS * NTOP * sizeof(int));
    _Float16* wf16 = (_Float16*)(ws + 4 * f16buf + (size_t)NB * NS * NTOP * sizeof(int)
                                 + (size_t)NB * NHW * sizeof(float));

    zero_counts_kernel<<<(NB * NHW) / 256, 256, 0, stream>>>(counts);
    wcvt_kernel<<<(3 * NC * NC) / 256, 256, 0, stream>>>(wq, wk, wv, wf16);
    ln_kernel<<<(NB * NHW) / 256, 256, 0, stream>>>(x, gamma, beta, xn);
    qkv_kernel<<<NB * (NHW / 128), 256, 0, stream>>>(xn, wf16, qws, kws, vws, out);
    topk_kernel<<<NB * NS, 256, 0, stream>>>(aff, idx, counts);
    attn_kernel<<<NB * NS * NHEAD, 256, 0, stream>>>(qws, kws, vws, idx, out);
    finalize_kernel<<<(size_t)NB * NC * NHW / 256, 256, 0, stream>>>(out, counts);
}
